// VolatilityLSTM_87024627351808
// MI455X (gfx1250) — matrix operand order repaired and checked by reading, not compiled, not run on hardware
//
#include <hip/hip_runtime.h>
#include <hip/hip_bf16.h>

typedef __bf16 bf16_t;
typedef __attribute__((ext_vector_type(16))) __bf16 v16bf;
typedef __attribute__((ext_vector_type(8)))  __bf16 v8bf;
typedef __attribute__((ext_vector_type(4)))  __bf16 v4bf;
typedef __attribute__((ext_vector_type(8)))  float  v8f;
typedef __attribute__((ext_vector_type(4)))  float  v4f;

constexpr int Bc = 256, Sc = 1024, Ic = 32, Hc = 128, Gc = 512;
constexpr int NHc = 8, HDc = 16, OUTc = 6;

// ---------------------------------------------------------------- converters
__global__ void f32_to_bf16_kernel(const float* __restrict__ src,
                                   bf16_t* __restrict__ dst, int n) {
  int i = blockIdx.x * blockDim.x + threadIdx.x;
  int stride = gridDim.x * blockDim.x;
  for (; i < n; i += stride) dst[i] = (bf16_t)src[i];
}

// Pack [w_ih | w_hh] -> bf16 [512][K_in+128] row-major; bias = b_ih + b_hh.
__global__ void pack_lstm_kernel(const float* __restrict__ w_ih,
                                 const float* __restrict__ w_hh,
                                 const float* __restrict__ b_ih,
                                 const float* __restrict__ b_hh,
                                 bf16_t* __restrict__ Wc, float* __restrict__ bc,
                                 int K_in) {
  const int K = K_in + Hc;
  const int total = Gc * K;
  const int i0 = blockIdx.x * blockDim.x + threadIdx.x;
  const int stride = gridDim.x * blockDim.x;
  for (int idx = i0; idx < total; idx += stride) {
    int n = idx / K, k = idx % K;
    float v = (k < K_in) ? w_ih[n * K_in + k] : w_hh[n * Hc + (k - K_in)];
    Wc[idx] = (bf16_t)v;
  }
  if (i0 < Gc) bc[i0] = b_ih[i0] + b_hh[i0];
}

// wk,wv rows (128..383) of in_proj_w -> bf16 [256][128]; bkv likewise.
__global__ void pack_kv_kernel(const float* __restrict__ in_w,
                               const float* __restrict__ in_b,
                               bf16_t* __restrict__ wkv, float* __restrict__ bkv) {
  int i = blockIdx.x * blockDim.x + threadIdx.x;
  if (i < 2 * Hc * Hc) wkv[i] = (bf16_t)in_w[Hc * Hc + i];
  if (i < 2 * Hc) bkv[i] = in_b[Hc + i];
}

// ---------------------------------------------------------------- LSTM layer
// One block = 16 batch rows, 512 threads = 16 waves. Wave w owns gate columns
// [32w, 32w+32): 2 WMMA N-tiles. Fused weights live in VGPRs for all 1024 steps.
template <int K_IN>
__global__ __launch_bounds__(512) void lstm_layer_kernel(
    const bf16_t* __restrict__ xin,  // [B][S][K_IN] bf16
    const bf16_t* __restrict__ Wc,   // [512][K_IN+128] bf16 row-major
    const float* __restrict__ bc,    // [512] combined bias
    bf16_t* __restrict__ hout) {     // [B][S][128] bf16
  constexpr int K = K_IN + Hc;   // GEMM inner dim
  constexpr int NK = K / 32;     // wmma k-steps (5 or 8)
  constexpr int MT = 16;         // batch tile

  extern __shared__ unsigned char smem_raw[];
  bf16_t* abuf = (bf16_t*)smem_raw;                                    // [MT][K]
  float* gates = (float*)(smem_raw + (size_t)MT * K * sizeof(bf16_t)); // [MT][512]

  const int tid = threadIdx.x;
  const int lane = tid & 31;             // wave32
  const int wave = tid >> 5;             // 0..15
  const int b0 = blockIdx.x * MT;
  const int lm = lane & 15;              // row/col within 16x16 tile
  const int lko = (lane >> 4) * 16;      // A-fragment K offset
  const int lkb = (lane >> 4) * 16;      // B-fragment K offset
  const int nbase = wave * 32;

  // Resident weight fragments (<= 8*2*8 = 128 VGPRs, reused 1024x).
  v16bf wfrag[NK][2];
#pragma unroll
  for (int ks = 0; ks < NK; ++ks)
#pragma unroll
    for (int ti = 0; ti < 2; ++ti) {
      int n = nbase + ti * 16 + lm;
      wfrag[ks][ti] = *(const v16bf*)(Wc + (size_t)n * K + ks * 32 + lkb);
    }
  const float bias0 = bc[nbase + lm];
  const float bias1 = bc[nbase + 16 + lm];

  // Per-thread cell state: 4 elements (m = wave, j = lane*4 .. +3).
  const int em = wave;
  const int ej = (tid & 31) * 4;
  float cst[4] = {0.f, 0.f, 0.f, 0.f};

  // h(t=-1) = 0 in abuf's h region (one b64 store).
  {
    v4bf z;
#pragma unroll
    for (int u = 0; u < 4; ++u) z[u] = (bf16_t)0.0f;
    *(v4bf*)(abuf + em * K + K_IN + ej) = z;
  }

  // x staging geometry: 4-bf16 chunks, one per thread (guard for layer 0).
  constexpr int CH = (MT * K_IN) / 4;     // 128 (K_IN=32) or 512 (K_IN=128)
  const int xm = tid / (K_IN / 4);
  const int xk = (tid % (K_IN / 4)) * 4;

  for (int t = 0; t < Sc; ++t) {
    // ---- stage x_t tile [16][K_IN] into abuf cols [0, K_IN)  (b64 copies)
    if (tid < CH)
      *(v4bf*)(abuf + xm * K + xk) =
          *(const v4bf*)(xin + ((size_t)(b0 + xm) * Sc + t) * K_IN + xk);
    if (t + 1 < Sc)  // prefetch next timestep's row -> global_prefetch_b8
      __builtin_prefetch(xin + ((size_t)(b0 + em) * Sc + (t + 1)) * K_IN, 0, 1);
    __syncthreads();

    // ---- gates(16x32 cols) = bias + [x_t | h] * Wc^T via WMMA bf16
    v8f acc0, acc1;
#pragma unroll
    for (int r = 0; r < 8; ++r) { acc0[r] = bias0; acc1[r] = bias1; }
#pragma unroll
    for (int ks = 0; ks < NK; ++ks) {
      const bf16_t* ap = abuf + lm * K + ks * 32 + lko;
      v8bf alo = *(const v8bf*)ap;
      v8bf ahi = *(const v8bf*)(ap + 8);
      v16bf a;
#pragma unroll
      for (int i = 0; i < 8; ++i) { a[i] = alo[i]; a[8 + i] = ahi[i]; }
      acc0 = __builtin_amdgcn_wmma_f32_16x16x32_bf16(false, a, false, wfrag[ks][0],
                                                     (short)0, acc0, false, false);
      acc1 = __builtin_amdgcn_wmma_f32_16x16x32_bf16(false, a, false, wfrag[ks][1],
                                                     (short)0, acc1, false, false);
    }
    // spill accumulators (C layout: VGPR r -> M = r + 8*(lane/16), N = lane%16)
#pragma unroll
    for (int r = 0; r < 8; ++r) {
      int m = r + (lane >> 4) * 8;
      gates[m * Gc + nbase + lm] = acc0[r];
      gates[m * Gc + nbase + 16 + lm] = acc1[r];
    }
    __syncthreads();

    // ---- elementwise cell update, 4 elems/thread; torch gate order i,f,g,o
    // four ds_load_b128, then one ds_store_b64 + one global_store_b64
    const float* gb = gates + em * Gc + ej;
    v4f vi = *(const v4f*)(gb);
    v4f vf = *(const v4f*)(gb + 128);
    v4f vg = *(const v4f*)(gb + 256);
    v4f vo = *(const v4f*)(gb + 384);
    v4bf hv4;
#pragma unroll
    for (int u = 0; u < 4; ++u) {
      float gi = 1.f / (1.f + __expf(-vi[u]));
      float gf = 1.f / (1.f + __expf(-vf[u]));
      float go = 1.f / (1.f + __expf(-vo[u]));
      float gg = tanhf(vg[u]);
      cst[u] = gf * cst[u] + gi * gg;
      hv4[u] = (bf16_t)(go * tanhf(cst[u]));
    }
    *(v4bf*)(abuf + em * K + K_IN + ej) = hv4;                       // next-step h
    *(v4bf*)(hout + ((size_t)(b0 + em) * Sc + t) * Hc + ej) = hv4;   // layer out
  }
}

// ---------------------------------------------------------------- K/V GEMM
// [B*S,128] x [256,128]^T + bias -> k (cols 0..127) and v (cols 128..255), bf16.
__global__ __launch_bounds__(256) void kv_gemm_kernel(
    const bf16_t* __restrict__ A, const bf16_t* __restrict__ Wkv,
    const float* __restrict__ bkv, bf16_t* __restrict__ kout,
    bf16_t* __restrict__ vout) {
  const int tid = threadIdx.x, lane = tid & 31;
  const int wg = blockIdx.x * 8 + (tid >> 5);  // global wave-tile id
  const int tm = wg >> 4;                      // 0..16383
  const int tn = wg & 15;                      // 0..15
  const int lm = lane & 15;
  const int lko = (lane >> 4) * 16, lkb = (lane >> 4) * 16;
  const int row = tm * 16 + lm;
  const int n = tn * 16 + lm;
  v8f acc;
  const float bv = bkv[n];
#pragma unroll
  for (int r = 0; r < 8; ++r) acc[r] = bv;
#pragma unroll
  for (int ks = 0; ks < 4; ++ks) {
    const bf16_t* ap = A + (size_t)row * 128 + ks * 32 + lko;
    v8bf alo = *(const v8bf*)ap;
    v8bf ahi = *(const v8bf*)(ap + 8);
    v16bf a;
#pragma unroll
    for (int i = 0; i < 8; ++i) { a[i] = alo[i]; a[8 + i] = ahi[i]; }
    v16bf b = *(const v16bf*)(Wkv + (size_t)n * 128 + ks * 32 + lkb);
    acc = __builtin_amdgcn_wmma_f32_16x16x32_bf16(false, a, false, b, (short)0,
                                                  acc, false, false);
  }
  // branchless epilogue: single pointer select instead of per-store exec masks
  bf16_t* outp = (tn < 8) ? (kout + (size_t)(tn * 16 + lm))
                          : (vout + (size_t)((tn - 8) * 16 + lm));
  const int mbase = tm * 16 + (lane >> 4) * 8;
#pragma unroll
  for (int r = 0; r < 8; ++r)
    outp[(size_t)(mbase + r) * 128] = (bf16_t)acc[r];
}

// ---------------------------------------------------------------- q (last token)
__global__ __launch_bounds__(128) void q_last_kernel(
    const bf16_t* __restrict__ h2, const float* __restrict__ in_w,
    const float* __restrict__ in_b, float* __restrict__ qbuf) {
  __shared__ float hl[Hc];
  const int b = blockIdx.x, n = threadIdx.x;
  hl[n] = (float)h2[((size_t)b * Sc + (Sc - 1)) * Hc + n];
  __syncthreads();
  float s = in_b[n];
  for (int e = 0; e < Hc; ++e) s += hl[e] * in_w[n * Hc + e];
  qbuf[b * Hc + n] = s;
}

// ---------------------------------------------------------------- attention
__global__ __launch_bounds__(128) void attn_kernel(
    const float* __restrict__ qbuf, const bf16_t* __restrict__ kbuf,
    const bf16_t* __restrict__ vbuf, float* __restrict__ ctxbuf) {
  __shared__ float red[128];
  __shared__ float qs[HDc];
  __shared__ float ctxl[HDc];
  const int b = blockIdx.x >> 3, hh = blockIdx.x & 7;
  const int tid = threadIdx.x;
  if (tid < HDc) { qs[tid] = qbuf[b * Hc + hh * HDc + tid]; ctxl[tid] = 0.f; }
  __syncthreads();

  float sc[8];
  float mx = -1e30f;
#pragma unroll
  for (int it = 0; it < 8; ++it) {
    const int s = it * 128 + tid;
    const bf16_t* kp = kbuf + ((size_t)b * Sc + s) * Hc + hh * HDc;
    float a = 0.f;
#pragma unroll
    for (int d = 0; d < HDc; ++d) a += qs[d] * (float)kp[d];
    sc[it] = a * 0.25f;  // 1/sqrt(16)
    mx = fmaxf(mx, sc[it]);
  }
  red[tid] = mx;
  __syncthreads();
  for (int off = 64; off > 0; off >>= 1) {
    if (tid < off) red[tid] = fmaxf(red[tid], red[tid + off]);
    __syncthreads();
  }
  mx = red[0];
  __syncthreads();

  float pacc[HDc];
#pragma unroll
  for (int d = 0; d < HDc; ++d) pacc[d] = 0.f;
  float ssum = 0.f;
#pragma unroll
  for (int it = 0; it < 8; ++it) {
    const int s = it * 128 + tid;
    const float w = __expf(sc[it] - mx);
    ssum += w;
    const bf16_t* vp = vbuf + ((size_t)b * Sc + s) * Hc + hh * HDc;
#pragma unroll
    for (int d = 0; d < HDc; ++d) pacc[d] += w * (float)vp[d];
  }
  red[tid] = ssum;
  __syncthreads();
  for (int off = 64; off > 0; off >>= 1) {
    if (tid < off) red[tid] += red[tid + off];
    __syncthreads();
  }
  const float inv = 1.f / red[0];
#pragma unroll
  for (int d = 0; d < HDc; ++d) atomicAdd(&ctxl[d], pacc[d]);  // ds_add_f32
  __syncthreads();
  if (tid < HDc) ctxbuf[b * Hc + hh * HDc + tid] = ctxl[tid] * inv;
}

// ---------------------------------------------------------------- MLP head
__global__ __launch_bounds__(128) void head_kernel(
    const float* __restrict__ ctxbuf, const float* __restrict__ ow,
    const float* __restrict__ ob, const float* __restrict__ d1w,
    const float* __restrict__ d1b, const float* __restrict__ d2w,
    const float* __restrict__ d2b, const float* __restrict__ d3w,
    const float* __restrict__ d3b, float* __restrict__ out) {
  __shared__ float cs[Hc], att[Hc], z1[64], z2[32];
  const int b = blockIdx.x, n = threadIdx.x;
  cs[n] = ctxbuf[b * Hc + n];
  __syncthreads();
  {
    float s = ob[n];
    for (int e = 0; e < Hc; ++e) s += cs[e] * ow[n * Hc + e];
    att[n] = s;
  }
  __syncthreads();
  if (n < 64) {
    float s = d1b[n];
    for (int e = 0; e < Hc; ++e) s += att[e] * d1w[n * Hc + e];
    z1[n] = fmaxf(s, 0.f);
  }
  __syncthreads();
  if (n < 32) {
    float s = d2b[n];
    for (int e = 0; e < 64; ++e) s += z1[e] * d2w[n * 64 + e];
    z2[n] = fmaxf(s, 0.f);
  }
  __syncthreads();
  if (n < OUTc) {
    float s = d3b[n];
    for (int e = 0; e < 32; ++e) s += z2[e] * d3w[n * 32 + e];
    out[b * OUTc + n] = s;
  }
}

// ---------------------------------------------------------------- launcher
extern "C" void kernel_launch(void* const* d_in, const int* in_sizes, int n_in,
                              void* d_out, int out_size, void* d_ws, size_t ws_size,
                              hipStream_t stream) {
  (void)in_sizes; (void)n_in; (void)out_size; (void)ws_size;
  const float* x = (const float*)d_in[0];
  const float* w_ih[3] = {(const float*)d_in[1], (const float*)d_in[5], (const float*)d_in[9]};
  const float* w_hh[3] = {(const float*)d_in[2], (const float*)d_in[6], (const float*)d_in[10]};
  const float* b_ih[3] = {(const float*)d_in[3], (const float*)d_in[7], (const float*)d_in[11]};
  const float* b_hh[3] = {(const float*)d_in[4], (const float*)d_in[8], (const float*)d_in[12]};
  const float* in_w = (const float*)d_in[13];
  const float* in_b = (const float*)d_in[14];
  const float* ow   = (const float*)d_in[15];
  const float* obv  = (const float*)d_in[16];
  const float* d1w = (const float*)d_in[17]; const float* d1b = (const float*)d_in[18];
  const float* d2w = (const float*)d_in[19]; const float* d2b = (const float*)d_in[20];
  const float* d3w = (const float*)d_in[21]; const float* d3b = (const float*)d_in[22];

  unsigned char* ws = (unsigned char*)d_ws;
  size_t off = 0;
  auto alloc = [&](size_t bytes) -> void* {
    void* p = ws + off;
    off = (off + bytes + 255) & ~(size_t)255;
    return p;
  };
  bf16_t* xbf = (bf16_t*)alloc((size_t)Bc * Sc * Ic * 2);
  bf16_t* h0  = (bf16_t*)alloc((size_t)Bc * Sc * Hc * 2);
  bf16_t* h1  = (bf16_t*)alloc((size_t)Bc * Sc * Hc * 2);
  bf16_t* h2  = (bf16_t*)alloc((size_t)Bc * Sc * Hc * 2);
  bf16_t* Wc0 = (bf16_t*)alloc((size_t)Gc * (Ic + Hc) * 2);
  bf16_t* Wc1 = (bf16_t*)alloc((size_t)Gc * (Hc + Hc) * 2);
  bf16_t* Wc2 = (bf16_t*)alloc((size_t)Gc * (Hc + Hc) * 2);
  float* bc0 = (float*)alloc(Gc * 4);
  float* bc1 = (float*)alloc(Gc * 4);
  float* bc2 = (float*)alloc(Gc * 4);
  bf16_t* wkv = (bf16_t*)alloc((size_t)2 * Hc * Hc * 2);
  float* bkv  = (float*)alloc(2 * Hc * 4);
  float* qbuf = (float*)alloc((size_t)Bc * Hc * 4);
  float* ctxb = (float*)alloc((size_t)Bc * Hc * 4);

  // 1. input activations -> bf16
  f32_to_bf16_kernel<<<1024, 256, 0, stream>>>(x, xbf, Bc * Sc * Ic);
  // 2. pack fused LSTM weights + biases; pack K/V weights
  pack_lstm_kernel<<<(Gc * (Ic + Hc) + 255) / 256, 256, 0, stream>>>(
      w_ih[0], w_hh[0], b_ih[0], b_hh[0], Wc0, bc0, Ic);
  pack_lstm_kernel<<<(Gc * (Hc + Hc) + 255) / 256, 256, 0, stream>>>(
      w_ih[1], w_hh[1], b_ih[1], b_hh[1], Wc1, bc1, Hc);
  pack_lstm_kernel<<<(Gc * (Hc + Hc) + 255) / 256, 256, 0, stream>>>(
      w_ih[2], w_hh[2], b_ih[2], b_hh[2], Wc2, bc2, Hc);
  pack_kv_kernel<<<(2 * Hc * Hc + 255) / 256, 256, 0, stream>>>(in_w, in_b, wkv, bkv);
  // 3. three recurrent layers (persistent, weights in VGPRs)
  const size_t sh0 = (size_t)16 * (Ic + Hc) * 2 + (size_t)16 * Gc * 4;  // ~37.9 KB
  const size_t sh1 = (size_t)16 * (Hc + Hc) * 2 + (size_t)16 * Gc * 4;  // 40 KB
  lstm_layer_kernel<Ic><<<Bc / 16, 512, sh0, stream>>>(xbf, Wc0, bc0, h0);
  lstm_layer_kernel<Hc><<<Bc / 16, 512, sh1, stream>>>(h0, Wc1, bc1, h1);
  lstm_layer_kernel<Hc><<<Bc / 16, 512, sh1, stream>>>(h1, Wc2, bc2, h2);
  // 4. K,V projections (reuse h0/h1 as K/V storage; their data is dead now)
  kv_gemm_kernel<<<(Bc * Sc / 16) * 16 / 8, 256, 0, stream>>>(h2, wkv, bkv, h0, h1);
  // 5. q for last token, attention, MLP head
  q_last_kernel<<<Bc, Hc, 0, stream>>>(h2, in_w, in_b, qbuf);
  attn_kernel<<<Bc * NHc, 128, 0, stream>>>(qbuf, h0, h1, ctxb);
  head_kernel<<<Bc, Hc, 0, stream>>>(ctxb, ow, obv, d1w, d1b, d2w, d2b, d3w, d3b,
                                     (float*)d_out);
}
